// ViT_32435593019955
// MI455X (gfx1250) — matrix operand order repaired and checked by reading, not compiled, not run on hardware
//
#include <hip/hip_runtime.h>
#include <hip/hip_bf16.h>
#include <math.h>

// ---------------------------------------------------------------------------
// Types for CDNA5 WMMA / TDM
// ---------------------------------------------------------------------------
typedef __bf16 bf16_t;
typedef __attribute__((ext_vector_type(16))) __bf16    v16bf;
typedef __attribute__((ext_vector_type(8)))  float     v8f;
typedef __attribute__((ext_vector_type(4)))  unsigned  u32x4;
typedef __attribute__((ext_vector_type(8)))  int       i32x8;
typedef __attribute__((ext_vector_type(4)))  int       i32x4;

#if defined(__has_builtin)
#if __has_builtin(__builtin_amdgcn_tensor_load_to_lds) && __has_builtin(__builtin_amdgcn_s_wait_tensorcnt)
#define HAVE_TDM 1
#endif
#endif

// Problem constants (match the reference)
#define LAYERS 12
#define EDIM   768
#define MFEAT  256
#define FDIM   3072
#define NPATCH 196
#define NTOK   197
#define BSZ    16
#define ROWS   (BSZ * NTOK)     // 3152 token rows
#define ROWSP  3200             // padded to multiple of 128
#define NCLS   1000
#define NCLSP  1024             // head weight physical column count
#define NPAD   224              // k_feat^T K padded (multiple of 32, zeros)

// GEMM tiling
#define BM 128
#define BN 128
#define BK 32
#define LDSS 40                 // LDS row stride (bf16 elems): 64B row + 16B pad

enum {
  F_BIAS    = 1,
  F_GELU    = 2,
  F_RESID   = 4,
  F_WRITEBF = 8,
  F_DIVROW  = 16,
  F_SUBROW  = 32,
  F_SKIPF32 = 64,
};

// ---------------------------------------------------------------------------
// Generic bf16 WMMA GEMM:  D[M,N] = A[M,K](bf16) @ B[K,N](bf16)  + epilogue
// Block: 256 threads = 8 waves. Block tile 128(M) x 128(N), K step 32.
// A tile staged via the Tensor Data Mover (one 32x128 2D descriptor with LDS
// padding reproducing the LDSS=40 layout) when the builtin is available.
// ALL staging loads are unconditional: callers guarantee physical padding.
// Nout guards the epilogue stores only.
// ---------------------------------------------------------------------------
__global__ __launch_bounds__(256)
void wmma_gemm_kernel(const bf16_t* __restrict__ A, const bf16_t* __restrict__ B,
                      float* Dout, bf16_t* Dbf, const float* residual,
                      const float* __restrict__ bias, const float* __restrict__ rowvec,
                      int Mrows, int N, int Nout, int K, int lda, int ldb, int ldd,
                      long long strideA, long long strideB, long long strideD,
                      long long strideRow, int flags)
{
  __shared__ bf16_t As[BM][LDSS];
  __shared__ bf16_t Bs[BN][LDSS];

  const int tid  = threadIdx.x;
  const int lane = tid & 31;
  const int wave = tid >> 5;
  const int half = lane >> 4;    // 0 or 1 (lane group)
  const int l15  = lane & 15;

  const int rowBase = blockIdx.y * BM;
  const int colBase = blockIdx.x * BN;
  const long long zb = blockIdx.z;

  A += zb * strideA;
  B += zb * strideB;
  if (Dout)     Dout     += zb * strideD;
  if (Dbf)      Dbf      += zb * strideD;
  if (residual) residual += zb * strideD;
  if (rowvec)   rowvec   += zb * strideRow;

  // cooperative-load assignments
  const int aRow  = tid >> 1;          // 0..127
  const int aKoff = (tid & 1) << 4;    // 0 or 16
  const int bKrow = tid >> 3;          // 0..31
  const int bNoff = (tid & 7) << 4;    // 0,16,..,112

  const bf16_t* Ald = A + (size_t)(rowBase + aRow) * lda + aKoff;
  const bf16_t* Bld = B + (size_t)bKrow * ldb + colBase + bNoff;

#if HAVE_TDM
  // TDM descriptor group1 (constant over the K loop): 2 byte elements,
  // pad_enable, pad_interval=16 DWORDs (one 64B tile row), pad_amount=4 DWORDs
  // (16B) -> LDS row stride = 80B = LDSS bf16 elements.  Tile 32 x 128,
  // tensor_dim0 = lda, dim0 stride = lda, tensor_dim1 = large (padded bufs).
  const unsigned td0 = (unsigned)lda;      // tensor_dim0 (elements)
  const unsigned td1 = 0x01000000u;        // tensor_dim1 (large)
  i32x8 g1;
  g1[0] = (int)((1u << 16) | (1u << 20) | (3u << 22) | (3u << 25));
  g1[1] = (int)((td0 & 0xFFFFu) << 16);
  g1[2] = (int)((td0 >> 16) | ((td1 & 0xFFFFu) << 16));
  g1[3] = (int)((td1 >> 16) | ((unsigned)BK << 16));          // tile_dim0 = 32
  g1[4] = (int)((unsigned)BM);                                 // tile_dim1 = 128
  g1[5] = (int)td0;                                            // dim0 stride lo
  g1[6] = 0;
  g1[7] = 0;
  const i32x4 gz = {0, 0, 0, 0};
  const unsigned lds_base = (unsigned)(size_t)(&As[0][0]);
  const unsigned long long gbase =
      (unsigned long long)(size_t)A + ((unsigned long long)rowBase * (unsigned)lda) * 2ull;
#endif

  v8f acc[8] = {};

  for (int kt = 0; kt < K; kt += BK) {
    // ---- stage A tile [BM x BK] ----
#if HAVE_TDM
    if (wave == 0) {
      const unsigned long long ga = gbase + (unsigned long long)kt * 2ull;
      u32x4 g0;
      g0[0] = 1u;                                    // count=1, user descriptor
      g0[1] = lds_base;                              // lds_addr (bytes)
      g0[2] = (unsigned)(ga & 0xFFFFFFFFull);        // global_addr[31:0]
      g0[3] = (unsigned)((ga >> 32) & 0x01FFFFFFull) | (2u << 30);  // addr hi | type=2
#if __clang_major__ >= 23
      const i32x8 g4 = {0, 0, 0, 0, 0, 0, 0, 0};
      __builtin_amdgcn_tensor_load_to_lds(g0, g1, gz, gz, g4, 0);
#else
      __builtin_amdgcn_tensor_load_to_lds(g0, g1, gz, gz, 0);
#endif
    }
#else
    {
      const uint4* src = (const uint4*)(Ald + kt);
      uint4* dst = (uint4*)&As[aRow][aKoff];
      dst[0] = src[0];
      dst[1] = src[1];
    }
#endif
    // ---- stage B tile [BK x BN] TRANSPOSED (Bs[n][k]): 32B per thread ----
    {
      const uint4 b0 = *(const uint4*)(Bld + (size_t)kt * ldb);
      const uint4 b1 = *(const uint4*)(Bld + (size_t)kt * ldb + 8);
      const bf16_t* p0 = (const bf16_t*)&b0;
      const bf16_t* p1 = (const bf16_t*)&b1;
#pragma unroll
      for (int j = 0; j < 8; ++j) Bs[bNoff + j][bKrow]     = p0[j];
#pragma unroll
      for (int j = 0; j < 8; ++j) Bs[bNoff + 8 + j][bKrow] = p1[j];
    }
#if HAVE_TDM
    if (wave == 0) __builtin_amdgcn_s_wait_tensorcnt(0);
#endif
    __syncthreads();

    // prefetch next K tile of B (speculative: dropped on fault, ISA 10.5)
    __builtin_prefetch(Bld + (size_t)(kt + BK) * ldb, 0, 1);
#if !HAVE_TDM
    __builtin_prefetch(Ald + kt + BK, 0, 1);
#endif

    // ---- A fragment (16x32 bf16): lane row = l&15; VGPR v holds K pair
    //      k = 16*(v>>2) + 8*(l>>4) + 2*(v&3)   (ISA 7.12.2) ----
    union F { v16bf v; unsigned u[8]; };
    F af, bg[2];
#pragma unroll
    for (int r = 0; r < 8; ++r) {
      const int k = ((r >> 2) << 4) + (half << 3) + ((r & 3) << 1);
      af.u[r] = *(const unsigned*)&As[(wave << 4) + l15][k];
    }
#pragma unroll
    for (int r = 0; r < 8; ++r) {
      const int k = ((r >> 2) << 4) + (half << 3) + ((r & 3) << 1);      // B: K pair = 16*(v>>2) + 8*half + 2*(v&3)
      bg[0].u[r] = *(const unsigned*)&Bs[l15][k];
    }

    // ---- 8 WMMA tiles across N, B fragments double-buffered ----
#pragma unroll
    for (int nt = 0; nt < 8; ++nt) {
      const int cur = nt & 1;
      if (nt < 7) {
#pragma unroll
        for (int r = 0; r < 8; ++r) {
          const int k = ((r >> 2) << 4) + (half << 3) + ((r & 3) << 1);
          bg[cur ^ 1].u[r] = *(const unsigned*)&Bs[((nt + 1) << 4) + l15][k];
        }
      }
      acc[nt] = __builtin_amdgcn_wmma_f32_16x16x32_bf16(
          /*neg_a=*/false, af.v, /*neg_b=*/false, bg[cur].v,
          /*c_mod=*/(short)0, acc[nt], /*reuse_a=*/false, /*reuse_b=*/false);
    }
    __syncthreads();
  }

  // ---- epilogue: D layout: N = lane&15 (+16*nt), M = v + 8*(lane>>4) ----
#pragma unroll
  for (int nt = 0; nt < 8; ++nt) {
#pragma unroll
    for (int i = 0; i < 8; ++i) {
      const int gm = rowBase + (wave << 4) + i + (half << 3);
      const int gn = colBase + (nt << 4) + l15;
      if (gm >= Mrows || gn >= Nout) continue;
      float v = acc[nt][i];
      if (flags & F_BIAS)   v += bias[gn];
      if (flags & F_SUBROW) v -= rowvec[gm];
      if (flags & F_DIVROW) v /= rowvec[gm];
      if (flags & F_GELU)   v = 0.5f * v * (1.0f + erff(v * 0.70710678118654752f));
      if (flags & F_RESID)  v += residual[(size_t)gm * ldd + gn];
      if (!(flags & F_SKIPF32)) Dout[(size_t)gm * ldd + gn] = v;
      if (flags & F_WRITEBF)    Dbf[(size_t)gm * ldd + gn] = (bf16_t)v;
    }
  }
}

// ---------------------------------------------------------------------------
// LayerNorm over E=768 (one block per row).  Optionally also emits the
// Performer query/key input xp = y * d^-0.25 (bf16) and diag = 0.5*||xp||^2.
// ---------------------------------------------------------------------------
__global__ __launch_bounds__(256)
void ln_kernel(const float* __restrict__ in, long long in_row_stride,
               const float* __restrict__ g, const float* __restrict__ b,
               bf16_t* out_bf, bf16_t* xp_bf, float* diag, float scale)
{
  __shared__ float sbuf[256];
  const int tid = threadIdx.x;
  const long long row = blockIdx.x;
  const float* x = in + row * in_row_stride;

  const float x0 = x[tid], x1 = x[tid + 256], x2 = x[tid + 512];

  sbuf[tid] = x0 + x1 + x2;
  __syncthreads();
  for (int o = 128; o > 0; o >>= 1) { if (tid < o) sbuf[tid] += sbuf[tid + o]; __syncthreads(); }
  const float mean = sbuf[0] * (1.0f / 768.0f);
  __syncthreads();

  sbuf[tid] = x0 * x0 + x1 * x1 + x2 * x2;
  __syncthreads();
  for (int o = 128; o > 0; o >>= 1) { if (tid < o) sbuf[tid] += sbuf[tid + o]; __syncthreads(); }
  const float var = sbuf[0] * (1.0f / 768.0f) - mean * mean;
  __syncthreads();

  const float rstd = rsqrtf(var + 1e-5f);
  const float y0 = (x0 - mean) * rstd * g[tid]       + b[tid];
  const float y1 = (x1 - mean) * rstd * g[tid + 256] + b[tid + 256];
  const float y2 = (x2 - mean) * rstd * g[tid + 512] + b[tid + 512];

  const size_t o768 = (size_t)row * 768;
  if (out_bf) {
    out_bf[o768 + tid]       = (bf16_t)y0;
    out_bf[o768 + tid + 256] = (bf16_t)y1;
    out_bf[o768 + tid + 512] = (bf16_t)y2;
  }
  if (xp_bf) {
    const float p0 = y0 * scale, p1 = y1 * scale, p2 = y2 * scale;
    xp_bf[o768 + tid]       = (bf16_t)p0;
    xp_bf[o768 + tid + 256] = (bf16_t)p1;
    xp_bf[o768 + tid + 512] = (bf16_t)p2;
    sbuf[tid] = p0 * p0 + p1 * p1 + p2 * p2;
    __syncthreads();
    for (int o = 128; o > 0; o >>= 1) { if (tid < o) sbuf[tid] += sbuf[tid + o]; __syncthreads(); }
    if (tid == 0) diag[row] = 0.5f * sbuf[0];
  }
}

// ---------------------------------------------------------------------------
// Performer helpers
// ---------------------------------------------------------------------------
__global__ __launch_bounds__(256)
void rowmax_kernel(const float* __restrict__ lg, float* rowmax)
{
  __shared__ float sbuf[256];
  const int tid = threadIdx.x;
  const size_t row = blockIdx.x;
  sbuf[tid] = lg[row * MFEAT + tid];
  __syncthreads();
  for (int o = 128; o > 0; o >>= 1) { if (tid < o) sbuf[tid] = fmaxf(sbuf[tid], sbuf[tid + o]); __syncthreads(); }
  if (tid == 0) rowmax[row] = sbuf[0];
}

__global__ __launch_bounds__(256)
void gmax_kernel(const float* __restrict__ rowmax, int n, float* gmax)
{
  __shared__ float sbuf[256];
  const int tid = threadIdx.x;
  float m = -3.402823466e38f;
  for (int i = tid; i < n; i += 256) m = fmaxf(m, rowmax[i]);
  sbuf[tid] = m;
  __syncthreads();
  for (int o = 128; o > 0; o >>= 1) { if (tid < o) sbuf[tid] = fmaxf(sbuf[tid], sbuf[tid + o]); __syncthreads(); }
  if (tid == 0) gmax[0] = sbuf[0];
}

// q_feat (row-major, f32 + bf16) and k_feat^T (feature-major [B][M][NPAD], bf16,
// zero padded n>=197 so the kv GEMM K loop is guard-free).
__global__ __launch_bounds__(256)
void feat_kernel(const float* __restrict__ lg, const float* __restrict__ rowmax,
                 const float* __restrict__ gmax,
                 float* qf32, bf16_t* qbf, bf16_t* ktbf)
{
  const int idx = blockIdx.x * 256 + threadIdx.x;   // over BSZ*MFEAT*NPAD
  if (idx >= BSZ * MFEAT * NPAD) return;
  const int n = idx % NPAD;
  const int m = (idx / NPAD) % MFEAT;
  const int b = idx / (NPAD * MFEAT);
  if (n < NTOK) {
    const size_t li = ((size_t)(b * NTOK + n)) * MFEAT + m;
    const float v  = lg[li];
    const float q  = expf(v - rowmax[b * NTOK + n]) * 0.0625f + 1e-4f;  // m^-0.5 = 1/16
    const float kf = expf(v - gmax[0])              * 0.0625f + 1e-4f;
    qf32[li] = q;
    qbf[li]  = (bf16_t)q;
    ktbf[idx] = (bf16_t)kf;
  } else {
    ktbf[idx] = (bf16_t)0.0f;
  }
}

__global__ __launch_bounds__(256)
void ksum_kernel(const bf16_t* __restrict__ ktbf, float* ksum)
{
  const int idx = blockIdx.x * 256 + threadIdx.x;   // over BSZ*MFEAT
  if (idx >= BSZ * MFEAT) return;
  const bf16_t* p = ktbf + (size_t)idx * NPAD;
  float s = 0.0f;
  for (int n = 0; n < NTOK; ++n) s += (float)p[n];
  ksum[idx] = s;
}

__global__ __launch_bounds__(256)
void den_kernel(const float* __restrict__ qf32, const float* __restrict__ ksum, float* den)
{
  const int idx = blockIdx.x * 256 + threadIdx.x;   // over ROWS
  if (idx >= ROWS) return;
  const int b = idx / NTOK;
  const float* q  = qf32 + (size_t)idx * MFEAT;
  const float* ks = ksum + b * MFEAT;
  float s = 0.0f;
  for (int m = 0; m < MFEAT; ++m) s += q[m] * ks[m];
  den[idx] = s;
}

// ---------------------------------------------------------------------------
// Data prep kernels
// ---------------------------------------------------------------------------
__global__ __launch_bounds__(256)
void im2col_kernel(const float* __restrict__ x, bf16_t* out)
{
  const int idx = blockIdx.x * 256 + threadIdx.x;   // over BSZ*NPATCH*768
  if (idx >= BSZ * NPATCH * 768) return;
  const int k = idx % 768;
  const int p = (idx / 768) % NPATCH;
  const int b = idx / (768 * NPATCH);
  const int c  = k >> 8;
  const int py = (k >> 4) & 15;
  const int px = k & 15;
  const int gy = p / 14, gx = p % 14;
  const size_t src = (((size_t)b * 3 + c) * 224 + gy * 16 + py) * 224 + gx * 16 + px;
  out[idx] = (bf16_t)x[src];
}

// h[b,0,:] = cls + pos[0];  h[b,n>0,:] += pos[n]
__global__ __launch_bounds__(256)
void finish_embed_kernel(const float* __restrict__ cls, const float* __restrict__ pos,
                         float* h)
{
  const int idx = blockIdx.x * 256 + threadIdx.x;   // over ROWS*768
  if (idx >= ROWS * 768) return;
  const int e = idx % 768;
  const int n = (idx / 768) % NTOK;
  float v;
  if (n == 0) v = cls[e] + pos[e];
  else        v = h[idx] + pos[(size_t)n * 768 + e];
  h[idx] = v;
}

__global__ __launch_bounds__(256)
void cvt_kernel(const float* __restrict__ in, bf16_t* out, size_t n)
{
  const size_t idx = (size_t)blockIdx.x * 256 + threadIdx.x;
  if (idx < n) out[idx] = (bf16_t)in[idx];
}

// head_w [768][1000] -> padded bf16 [768][1024] (zeros in pad columns)
__global__ __launch_bounds__(256)
void cvt_headw_kernel(const float* __restrict__ in, bf16_t* out)
{
  const int idx = blockIdx.x * 256 + threadIdx.x;   // over 768*NCLSP
  if (idx >= 768 * NCLSP) return;
  const int c = idx % NCLSP;
  const int r = idx / NCLSP;
  out[idx] = (c < NCLS) ? (bf16_t)in[(size_t)r * NCLS + c] : (bf16_t)0.0f;
}

// in: [z][R][C] -> out: [z][C][R], converted to bf16
__global__ __launch_bounds__(256)
void tcvt_kernel(const float* __restrict__ in, bf16_t* out, int R, int C)
{
  const size_t base = (size_t)blockIdx.z * R * C;
  const int idx = blockIdx.x * 256 + threadIdx.x;
  if (idx >= R * C) return;
  const int r = idx / C, c = idx % C;
  out[base + (size_t)c * R + r] = (bf16_t)in[base + idx];
}

__global__ __launch_bounds__(256)
void zero_bf16_kernel(bf16_t* p, size_t n)
{
  const size_t idx = (size_t)blockIdx.x * 256 + threadIdx.x;
  if (idx < n) p[idx] = (bf16_t)0.0f;
}

// ---------------------------------------------------------------------------
// Host launcher
// ---------------------------------------------------------------------------
static inline void gemm(hipStream_t stream,
                        const bf16_t* A, const bf16_t* B, float* Dout, bf16_t* Dbf,
                        const float* resid, const float* bias, const float* rowvec,
                        int M, int N, int Nout, int K, int lda, int ldb, int ldd,
                        long long sA, long long sB, long long sD, long long sR,
                        int batches, int flags)
{
  dim3 grid((N + BN - 1) / BN, (M + BM - 1) / BM, batches);
  wmma_gemm_kernel<<<grid, 256, 0, stream>>>(A, B, Dout, Dbf, resid, bias, rowvec,
                                             M, N, Nout, K, lda, ldb, ldd, sA, sB, sD, sR, flags);
}

extern "C" void kernel_launch(void* const* d_in, const int* in_sizes, int n_in,
                              void* d_out, int out_size, void* d_ws, size_t ws_size,
                              hipStream_t stream)
{
  const float* x       = (const float*)d_in[0];
  const float* cls_tok = (const float*)d_in[1];
  const float* pos_emb = (const float*)d_in[2];
  const float* patch_w = (const float*)d_in[3];
  const float* patch_b = (const float*)d_in[4];
  const float* ln1_g   = (const float*)d_in[5];
  const float* ln1_b   = (const float*)d_in[6];
  const float* proj    = (const float*)d_in[7];
  const float* ln2_g   = (const float*)d_in[8];
  const float* ln2_b   = (const float*)d_in[9];
  const float* w1      = (const float*)d_in[10];
  const float* b1      = (const float*)d_in[11];
  const float* w2      = (const float*)d_in[12];
  const float* b2      = (const float*)d_in[13];
  const float* lnf_g   = (const float*)d_in[14];
  const float* lnf_b   = (const float*)d_in[15];
  const float* head_w  = (const float*)d_in[16];
  const float* head_b  = (const float*)d_in[17];
  float* out = (float*)d_out;
  (void)in_sizes; (void)n_in; (void)out_size; (void)ws_size;

  // ---- workspace carve-out (256B aligned) ----
  // All A-side bf16 buffers are over-allocated so the guard-free staging
  // (vector loads or TDM tiles) can overrun rows; overrun rows only feed
  // epilogue-masked output rows.
  char* wp = (char*)d_ws;
  auto alloc = [&](size_t bytes) -> void* {
    void* p = (void*)wp;
    wp += (bytes + 255) & ~(size_t)255;
    return p;
  };
  bf16_t* wpeT_bf  = (bf16_t*)alloc((size_t)768 * 768 * 2);                 // [K=768][E=768]
  bf16_t* projT_bf = (bf16_t*)alloc((size_t)LAYERS * 768 * MFEAT * 2);      // [L][E][M]
  bf16_t* w1_bf    = (bf16_t*)alloc((size_t)LAYERS * 768 * FDIM * 2);
  bf16_t* w2_bf    = (bf16_t*)alloc((size_t)LAYERS * FDIM * 768 * 2);
  bf16_t* headw_bf = (bf16_t*)alloc((size_t)768 * NCLSP * 2);               // padded cols
  bf16_t* pat_bf   = (bf16_t*)alloc(((size_t)BSZ * NPATCH + 128) * 768 * 2);
  float*  h        = (float*)alloc((size_t)ROWS * 768 * 4);
  bf16_t* xln_bf   = (bf16_t*)alloc((size_t)ROWSP * 768 * 2);
  bf16_t* xp_bf    = (bf16_t*)alloc((size_t)ROWSP * 768 * 2);
  float*  diag     = (float*)alloc((size_t)ROWS * 4);
  float*  lg       = (float*)alloc((size_t)ROWS * MFEAT * 4);
  float*  rowmax   = (float*)alloc((size_t)ROWS * 4);
  float*  gmax     = (float*)alloc(256);
  float*  qf32     = (float*)alloc((size_t)ROWS * MFEAT * 4);
  bf16_t* q_bf     = (bf16_t*)alloc(((size_t)ROWS + 128) * MFEAT * 2);
  bf16_t* kt_bf    = (bf16_t*)alloc((size_t)BSZ * MFEAT * NPAD * 2);
  float*  ksum     = (float*)alloc((size_t)BSZ * MFEAT * 4);
  float*  den      = (float*)alloc((size_t)ROWS * 4);
  bf16_t* kv_bf    = (bf16_t*)alloc((size_t)BSZ * MFEAT * 768 * 2);
  bf16_t* min_bf   = (bf16_t*)alloc((size_t)ROWSP * 768 * 2);
  bf16_t* t_bf     = (bf16_t*)alloc((size_t)ROWSP * FDIM * 2);
  bf16_t* cls_bf   = (bf16_t*)alloc((size_t)128 * 768 * 2);

  const float xscale = (float)pow(768.0, -0.25);   // d^{-1/4}

  auto blocks = [](size_t n) { return (unsigned)((n + 255) / 256); };

  // ---- weight conversion (deterministic, every call) ----
  cvt_kernel<<<blocks((size_t)LAYERS * 768 * FDIM), 256, 0, stream>>>(w1, w1_bf, (size_t)LAYERS * 768 * FDIM);
  cvt_kernel<<<blocks((size_t)LAYERS * FDIM * 768), 256, 0, stream>>>(w2, w2_bf, (size_t)LAYERS * FDIM * 768);
  cvt_headw_kernel<<<blocks((size_t)768 * NCLSP), 256, 0, stream>>>(head_w, headw_bf);
  { dim3 g(blocks((size_t)768 * 768), 1, 1);
    tcvt_kernel<<<g, 256, 0, stream>>>(patch_w, wpeT_bf, 768, 768); }            // [E][CPP]->[CPP][E]
  { dim3 g(blocks((size_t)MFEAT * 768), 1, LAYERS);
    tcvt_kernel<<<g, 256, 0, stream>>>(proj, projT_bf, MFEAT, 768); }            // [M][E]->[E][M]

  // Zero xln slack rows (consumed as B-side K-overrun rows in the kv GEMM;
  // multiplied by zero A there, but keep them finite-zero for cleanliness).
  zero_bf16_kernel<<<blocks((size_t)(ROWSP - ROWS) * 768), 256, 0, stream>>>(
      xln_bf + (size_t)ROWS * 768, (size_t)(ROWSP - ROWS) * 768);

  // ---- patch embedding ----
  im2col_kernel<<<blocks((size_t)BSZ * NPATCH * 768), 256, 0, stream>>>(x, pat_bf);
  // tok = patches @ Wpe + b  -> written into h rows 1..196 of each batch
  gemm(stream, pat_bf, wpeT_bf, h + 768, nullptr, nullptr, patch_b, nullptr,
       NPATCH, 768, 768, 768, 768, 768, 768,
       (long long)NPATCH * 768, 0, (long long)NTOK * 768, 0, BSZ, F_BIAS);
  finish_embed_kernel<<<blocks((size_t)ROWS * 768), 256, 0, stream>>>(cls_tok, pos_emb, h);

  // ---- transformer layers ----
  for (int l = 0; l < LAYERS; ++l) {
    // LN1: xln (bf16, performer values), xp = xln * d^-1/4 (bf16), diag
    ln_kernel<<<ROWS, 256, 0, stream>>>(h, 768, ln1_g + l * 768, ln1_b + l * 768,
                                        xln_bf, xp_bf, diag, xscale);

    // logits - diag : [3152 x 256] = xp @ projT
    gemm(stream, xp_bf, projT_bf + (size_t)l * 768 * MFEAT, lg, nullptr, nullptr,
         nullptr, diag, ROWS, MFEAT, MFEAT, 768, 768, MFEAT, MFEAT,
         0, 0, 0, 0, 1, F_SUBROW);

    rowmax_kernel<<<ROWS, 256, 0, stream>>>(lg, rowmax);
    gmax_kernel<<<1, 256, 0, stream>>>(rowmax, ROWS, gmax);
    feat_kernel<<<blocks((size_t)BSZ * MFEAT * NPAD), 256, 0, stream>>>(lg, rowmax, gmax,
                                                                        qf32, q_bf, kt_bf);
    ksum_kernel<<<blocks((size_t)BSZ * MFEAT), 256, 0, stream>>>(kt_bf, ksum);
    den_kernel<<<blocks((size_t)ROWS), 256, 0, stream>>>(qf32, ksum, den);

    // kv[b] = k_feat^T @ xln : [256 x 768], K padded to 224 (zeros in A pad)
    gemm(stream, kt_bf, xln_bf, nullptr, kv_bf, nullptr, nullptr, nullptr,
         MFEAT, 768, 768, NPAD, NPAD, 768, 768,
         (long long)MFEAT * NPAD, (long long)NTOK * 768, (long long)MFEAT * 768, 0,
         BSZ, F_WRITEBF | F_SKIPF32);

    // h += (q_feat @ kv) / den   (in-place residual)
    gemm(stream, q_bf, kv_bf, h, nullptr, h, nullptr, den,
         NTOK, 768, 768, MFEAT, MFEAT, 768, 768,
         (long long)NTOK * MFEAT, (long long)MFEAT * 768, (long long)NTOK * 768,
         (long long)NTOK, BSZ, F_DIVROW | F_RESID);

    // LN2 -> m_in (bf16)
    ln_kernel<<<ROWS, 256, 0, stream>>>(h, 768, ln2_g + l * 768, ln2_b + l * 768,
                                        min_bf, nullptr, nullptr, 1.0f);

    // t = gelu(m_in @ W1 + b1)  (bf16 only)
    gemm(stream, min_bf, w1_bf + (size_t)l * 768 * FDIM, nullptr, t_bf, nullptr,
         b1 + l * FDIM, nullptr, ROWS, FDIM, FDIM, 768, 768, FDIM, FDIM,
         0, 0, 0, 0, 1, F_BIAS | F_GELU | F_WRITEBF | F_SKIPF32);

    // h += t @ W2 + b2
    gemm(stream, t_bf, w2_bf + (size_t)l * FDIM * 768, h, nullptr, h,
         b2 + l * 768, nullptr, ROWS, 768, 768, FDIM, FDIM, 768, 768,
         0, 0, 0, 0, 1, F_BIAS | F_RESID);
  }

  // ---- final LN on cls rows (token 0 of each batch) ----
  ln_kernel<<<BSZ, 256, 0, stream>>>(h, (long long)NTOK * 768, lnf_g, lnf_b,
                                     cls_bf, nullptr, nullptr, 1.0f);

  // ---- classifier head: [16 x 1000] (loads use padded 1024 columns) ----
  gemm(stream, cls_bf, headw_bf, out, nullptr, nullptr, head_b, nullptr,
       BSZ, NCLSP, NCLS, 768, 768, NCLSP, NCLS, 0, 0, 0, 0, 1, F_BIAS);
}
